// SpectralGraphConv_83150566851023
// MI455X (gfx1250) — matrix operand order repaired and checked by reading, not compiled, not run on hardware
//
#include <hip/hip_runtime.h>

typedef _Float16 v16h __attribute__((ext_vector_type(16)));
typedef _Float16 v8h  __attribute__((ext_vector_type(8)));
typedef float    v8f  __attribute__((ext_vector_type(8)));

#define B_    32
#define C_    16
#define N_    64
#define T_    4096
#define OUT_  16
#define ROWS_ 1024          // OUT_ * N_
#define TT    128           // t-tile width per block
#define LSTR  72            // LDS row stride in f16 (144 B: 16B-aligned, bank-friendly)

// ---------------------------------------------------------------------------
// Kernel 1: K[(o,i), j] = sum_n U[i,n] * theta[n,o] * U[j,n]  -> f16 in d_ws
// ---------------------------------------------------------------------------
__global__ __launch_bounds__(64)
void build_K_kernel(const float* __restrict__ U, const float* __restrict__ theta,
                    _Float16* __restrict__ Kh) {
  const int r = blockIdx.x;        // row = o*64 + i, 0..1023
  const int j = threadIdx.x;       // 0..63
  const int o = r >> 6;
  const int i = r & 63;
  float s = 0.f;
  #pragma unroll 8
  for (int n = 0; n < N_; ++n)
    s += U[i * N_ + n] * theta[n * OUT_ + o] * U[j * N_ + n];
  Kh[r * N_ + j] = (_Float16)s;
}

static __device__ __forceinline__ v16h cat8(v8h lo, v8h hi) {
  return __builtin_shufflevector(lo, hi, 0, 1, 2, 3, 4, 5, 6, 7,
                                 8, 9, 10, 11, 12, 13, 14, 15);
}

// ---------------------------------------------------------------------------
// Kernel 2: per (b, t-tile):
//   Stage A: xs[j,t] = sum_c x[b,c,j,t]  (f32 accumulate, f16 transposed LDS)
//   Stage B: y[b,row,t] = K(1024x64) @ xs(64xTT) via v_wmma_f32_16x16x32_f16
// ---------------------------------------------------------------------------
__global__ __launch_bounds__(256)
void spectral_main_kernel(const float* __restrict__ x,
                          const _Float16* __restrict__ Kh,
                          float* __restrict__ y) {
  __shared__ __align__(16) _Float16 xsT[TT][LSTR];   // [t][k], 18.4 KB

  const int b   = blockIdx.y;
  const int t0  = blockIdx.x * TT;
  const int tid = threadIdx.x;

  // ---- Stage A: channel reduction, coalesced float4 streams -> LDS (f16, transposed)
  for (int e = tid; e < N_ * (TT / 4); e += 256) {
    const int j  = e >> 5;           // 0..63
    const int t4 = (e & 31) << 2;    // 0..124 step 4
    const float* px = x + (size_t)b * C_ * N_ * T_ + (size_t)j * T_ + t0 + t4;
    float4 a = make_float4(0.f, 0.f, 0.f, 0.f);
    #pragma unroll
    for (int c = 0; c < C_; ++c) {
      float4 v = *(const float4*)(px + (size_t)c * N_ * T_);
      a.x += v.x; a.y += v.y; a.z += v.z; a.w += v.w;
    }
    xsT[t4 + 0][j] = (_Float16)a.x;
    xsT[t4 + 1][j] = (_Float16)a.y;
    xsT[t4 + 2][j] = (_Float16)a.z;
    xsT[t4 + 3][j] = (_Float16)a.w;
  }
  __syncthreads();

  // ---- Stage B
  const int wave   = tid >> 5;
  const int lane   = tid & 31;
  const int lane_n = lane & 15;   // row-in-tile (A) / col-in-tile (B, C/D)
  const int lane_h = lane >> 4;   // half-wave selector

  // Preload A fragments (this wave's 8 row-tiles x 2 k-steps) into registers.
  // A k order used here (contiguous, matching B): lanes 0-15 hold K 0-15; lanes 16-31 hold K 16-31.
  v16h afrag[8][2];
  #pragma unroll
  for (int q = 0; q < 8; ++q) {
    const int row = (wave * 8 + q) * 16 + lane_n;
    #pragma unroll
    for (int ks = 0; ks < 2; ++ks) {
      const _Float16* p = Kh + row * N_ + ks * 32 + lane_h * 16;
      afrag[q][ks] = cat8(*(const v8h*)p, *(const v8h*)(p + 8));
    }
  }

  for (int ct = 0; ct < TT / 16; ++ct) {
    // B fragments from LDS: contiguous k order as in A, lanes 0-15 hold K 0-15, lanes 16-31 hold K 16-31.
    // Per-lane: 16 contiguous f16 (two b128 loads), conflict-free banking.
    const _Float16* qb = &xsT[ct * 16 + lane_n][lane_h * 16];
    const v16h b0 = cat8(*(const v8h*)(qb),      *(const v8h*)(qb + 8));
    const v16h b1 = cat8(*(const v8h*)(qb + 32), *(const v8h*)(qb + 40));

    #pragma unroll
    for (int q = 0; q < 8; ++q) {
      v8f acc = {};
      acc = __builtin_amdgcn_wmma_f32_16x16x32_f16(false, afrag[q][0], false, b0,
                                                   (short)0, acc, false, false);
      acc = __builtin_amdgcn_wmma_f32_16x16x32_f16(false, afrag[q][1], false, b1,
                                                   (short)0, acc, false, false);
      // C/D f32 16x16 layout: VGPR v -> row v (lanes 0-15) and row v+8 (lanes 16-31)
      float* yp = y + ((size_t)b * ROWS_ + (size_t)(wave * 8 + q) * 16 + 8 * lane_h) * T_
                    + t0 + ct * 16 + lane_n;
      #pragma unroll
      for (int v = 0; v < 8; ++v) yp[(size_t)v * T_] = acc[v];
    }
  }
}

// ---------------------------------------------------------------------------
extern "C" void kernel_launch(void* const* d_in, const int* in_sizes, int n_in,
                              void* d_out, int out_size, void* d_ws, size_t ws_size,
                              hipStream_t stream) {
  (void)in_sizes; (void)n_in; (void)out_size; (void)ws_size;
  const float* x     = (const float*)d_in[0];
  const float* U     = (const float*)d_in[1];
  const float* theta = (const float*)d_in[2];
  float*     y  = (float*)d_out;
  _Float16*  Kh = (_Float16*)d_ws;          // 1024*64*2 = 128 KB scratch

  build_K_kernel<<<dim3(ROWS_), dim3(64), 0, stream>>>(U, theta, Kh);
  spectral_main_kernel<<<dim3(T_ / TT, B_), dim3(256), 0, stream>>>(x, Kh, y);
}
